// HydraChannelMixer_51866025066916
// MI455X (gfx1250) — matrix operand order repaired and checked by reading, not compiled, not run on hardware
//
#include <hip/hip_runtime.h>
#include <hip/hip_bf16.h>
#include <math.h>

// ---------------------------------------------------------------------------
// HydraChannelMixer for MI455X (gfx1250, wave32, WMMA f32_16x16x32_f16)
// B=64, C=21, P=96, D=256, R=32, H=64.  Grid = B*P = 6144 blocks of 128 thr.
// Memory-bound: ~264 MB HBM traffic -> ~11.3 us floor @ 23.3 TB/s.
// Matmuls (Wd, Wq/Wk/Wv/Wg, Wu) run on the WMMA pipe in f16; gate MLPs on VALU.
// Slab gather uses GLOBAL_LOAD_ASYNC_TO_LDS_B128 (ASYNCcnt) when available.
// ---------------------------------------------------------------------------

typedef _Float16 half_t;
typedef __attribute__((ext_vector_type(16))) _Float16 v16h;
typedef __attribute__((ext_vector_type(8)))  float    v8f;
typedef int v4i __attribute__((ext_vector_type(4)));
typedef __attribute__((address_space(1))) v4i* as1_v4i_p;
typedef __attribute__((address_space(3))) v4i* as3_v4i_p;

#ifndef __has_builtin
#define __has_builtin(x) 0
#endif
#if __has_builtin(__builtin_amdgcn_global_load_async_to_lds_b128) && \
    __has_builtin(__builtin_amdgcn_s_wait_asynccnt)
#define USE_ASYNC_LDS 1
#else
#define USE_ASYNC_LDS 0
#endif

#define BDIM 128
#define C_   21
#define P_   96
#define D_   256
#define R_   32
#define H_   64

// workspace layout (bytes)
#define WS_WDT   0        // half[32*256]  WdT [N=32][K=256]
#define WS_WQT   16384    // half[32*32]
#define WS_WKT   18432
#define WS_WVT   20480
#define WS_WGT   22528
#define WS_WUT   24576    // half[256*32]  WuT [N=256][K=32]
#define WS_CHL   40960    // float[256] channel logits

__device__ __forceinline__ float gelu_exact(float v) {
    return 0.5f * v * (1.0f + erff(v * 0.70710678118654752f));
}
__device__ __forceinline__ float sigm(float v) {
    return 1.0f / (1.0f + __expf(-v));
}

// A-matrix 16x32 f16 fragment from LDS (row-major, stride halves).
// ISA 7.12.2: lane m = lane&15; VGPR v holds K = (v<4?0:16) + (lane>=16?8:0) + 2*(v&3) + t
__device__ __forceinline__ v16h load_a_lds(const half_t* base, int m0, int k0, int stride) {
    const int lane = threadIdx.x & 31;
    const int m = m0 + (lane & 15);
    const int kg = (lane >> 4) ? 8 : 0;
    const half_t* row = base + m * stride + k0 + kg;
    union { v16h h; unsigned int u[8]; } f;
#pragma unroll
    for (int v = 0; v < 8; ++v) {
        const int k = ((v & 4) ? 16 : 0) + 2 * (v & 3);
        f.u[v] = *(const unsigned int*)(row + k);
    }
    return f.h;
}

// B-matrix 32x16 f16 fragment from global, weight stored transposed [N][K].
// VGPR v holds K = (v<4?0:16) + (lane>=16?8:0) + 2*(v&3) + t (N=lane&15) — 2 halves/VGPR.
__device__ __forceinline__ v16h load_b_glb(const half_t* Wt, int n0, int k0, int K) {
    const int lane = threadIdx.x & 31;
    const int n = n0 + (lane & 15);
    const int kg = (lane >> 4) ? 8 : 0;
    const half_t* row = Wt + n * K + k0 + kg;
    union { v16h h; unsigned int u[8]; } f;
#pragma unroll
    for (int v = 0; v < 8; ++v) f.u[v] = *(const unsigned int*)(row + ((v & 4) ? 16 : 0) + 2 * (v & 3));
    return f.h;
}

__device__ __forceinline__ v8f wmma_f16(v16h a, v16h b, v8f c) {
    return __builtin_amdgcn_wmma_f32_16x16x32_f16(false, a, false, b, (short)0, c, false, false);
}

// ---------------------------------------------------------------------------
// Prep kernel: f32 -> f16 transposed weights + constant channel logits.
// ---------------------------------------------------------------------------
__global__ __launch_bounds__(256) void hydra_prep(
    const float* __restrict__ Wd, const float* __restrict__ Wq,
    const float* __restrict__ Wk, const float* __restrict__ Wv,
    const float* __restrict__ Wg, const float* __restrict__ Wu,
    const float* __restrict__ cn1w, const float* __restrict__ cn1b,
    const float* __restrict__ cn2w, const float* __restrict__ cn2b,
    char* __restrict__ ws)
{
    const int t = threadIdx.x;
    half_t* WdT = (half_t*)(ws + WS_WDT);
    half_t* WqT = (half_t*)(ws + WS_WQT);
    half_t* WkT = (half_t*)(ws + WS_WKT);
    half_t* WvT = (half_t*)(ws + WS_WVT);
    half_t* WgT = (half_t*)(ws + WS_WGT);
    half_t* WuT = (half_t*)(ws + WS_WUT);
    float*  chl = (float*)(ws + WS_CHL);

    // Wd [K=256][N=32] -> WdT [N=32][K=256]
    for (int o = t; o < 32 * 256; o += 256) {
        const int n = o >> 8, k = o & 255;
        WdT[o] = (half_t)Wd[k * 32 + n];
    }
    // R x R weights -> [N=32][K=32]
    for (int o = t; o < 32 * 32; o += 256) {
        const int n = o >> 5, k = o & 31;
        WqT[o] = (half_t)Wq[k * 32 + n];
        WkT[o] = (half_t)Wk[k * 32 + n];
        WvT[o] = (half_t)Wv[k * 32 + n];
        WgT[o] = (half_t)Wg[k * 32 + n];
    }
    // Wu [K=32][N=256] -> WuT [N=256][K=32]
    for (int o = t; o < 256 * 32; o += 256) {
        const int n = o >> 5, k = o & 31;
        WuT[o] = (half_t)Wu[k * 256 + n];
    }
    // channel logits: gelu(lc @ cn1_w + cn1_b) @ cn2_w + cn2_b  (constant)
    __shared__ float hid[H_];
    const float lc = logf(21.0f) / logf(1000.0f);
    if (t < H_) hid[t] = gelu_exact(lc * cn1w[t] + cn1b[t]);
    __syncthreads();
    if (t < D_) {
        float a = cn2b[t];
#pragma unroll 8
        for (int j = 0; j < H_; ++j) a += hid[j] * cn2w[j * D_ + t];
        chl[t] = a;
    }
}

// ---------------------------------------------------------------------------
// Main kernel: one block per (b,p) slab of [C=21, D=256].
// ---------------------------------------------------------------------------
#define SH 260   // h stride (floats);  260*4 = 1040 B = 65*16 -> rows 16B aligned
#define SN 258   // h_norm stride (halves)
#define OFF_H      0
#define OFF_R1     21840   // h_norm (32*258 half) aliased with QKVG (4*32*33 f32)
#define OFF_HLOW   38736   // float[32*33]
#define OFF_HLOW16 42960   // half[32*34]
#define OFF_A2     45136   // half[32*34]
#define OFF_VAR    47312   // float[256]
#define OFF_GATE   48336   // float[256]
#define OFF_HID    49360   // float[64]
#define OFF_QINV   49616   // float[32]
#define OFF_KINV   49744   // float[32]
#define OFF_GF     49872   // float[32]
#define SMEM_BYTES 50048

__global__ __launch_bounds__(BDIM) void hydra_main(
    const float* __restrict__ x,
    const float* __restrict__ ln_w, const float* __restrict__ ln_b,
    const float* __restrict__ bd,   const float* __restrict__ bg,
    const float* __restrict__ bu,
    const float* __restrict__ dn1w, const float* __restrict__ dn1b,
    const float* __restrict__ dn2w, const float* __restrict__ dn2b,
    const float* __restrict__ eps_p,
    const char* __restrict__ ws,
    float* __restrict__ out)
{
    __shared__ __align__(16) char smem[SMEM_BYTES];
    float*  sh_h     = (float*) (smem + OFF_H);
    half_t* sh_hn    = (half_t*)(smem + OFF_R1);
    float*  sh_qkvg  = (float*) (smem + OFF_R1);      // aliases sh_hn (disjoint lifetime)
    float*  sh_hlow  = (float*) (smem + OFF_HLOW);
    half_t* sh_hlow16= (half_t*)(smem + OFF_HLOW16);
    half_t* sh_a2    = (half_t*)(smem + OFF_A2);
    float*  sh_var   = (float*) (smem + OFF_VAR);
    float*  sh_gate  = (float*) (smem + OFF_GATE);
    float*  sh_hid   = (float*) (smem + OFF_HID);
    float*  sh_qinv  = (float*) (smem + OFF_QINV);
    float*  sh_kinv  = (float*) (smem + OFF_KINV);
    float*  sh_gf    = (float*) (smem + OFF_GF);

    const half_t* WdT = (const half_t*)(ws + WS_WDT);
    const half_t* WqT = (const half_t*)(ws + WS_WQT);
    const half_t* WkT = (const half_t*)(ws + WS_WKT);
    const half_t* WvT = (const half_t*)(ws + WS_WVT);
    const half_t* WgT = (const half_t*)(ws + WS_WGT);
    const half_t* WuT = (const half_t*)(ws + WS_WUT);
    const float*  chl = (const float*)(ws + WS_CHL);

    const int bp   = blockIdx.x;            // 0..B*P-1
    const int b    = bp / P_;
    const int p    = bp - b * P_;
    const int t    = threadIdx.x;
    const int lane = t & 31;
    const int wave = t >> 5;

    // ---- 1. gather slab h[c][d] = x[(b*C + c)*P + p][d] ---------------------
#if USE_ASYNC_LDS
    // HBM -> LDS directly (no VGPR round trip), 512 B per wave-instruction,
    // tracked by ASYNCcnt.  Each lane moves 2x16B per row.
    for (int c = wave; c < C_; c += 4) {
        const float* src = x + ((size_t)(b * C_ + c) * P_ + p) * D_;
        float* dst = sh_h + c * SH;
#pragma unroll
        for (int hh = 0; hh < 2; ++hh) {
            const int d = lane * 4 + hh * 128;
            __builtin_amdgcn_global_load_async_to_lds_b128(
                (as1_v4i_p)(uintptr_t)(src + d),
                (as3_v4i_p)(unsigned int)(uintptr_t)(dst + d),
                0, 0);
        }
    }
    __builtin_amdgcn_s_wait_asynccnt(0);
#else
    for (int c = wave; c < C_; c += 4) {
        const float4* src = (const float4*)(x + ((size_t)(b * C_ + c) * P_ + p) * D_);
        float4* dst = (float4*)(sh_h + c * SH);
#pragma unroll
        for (int q = lane; q < D_ / 4; q += 32) dst[q] = src[q];
    }
#endif
    __syncthreads();

    // ---- 2. LayerNorm over D per row (one wave per row, shuffle reduce) ----
    for (int c = wave; c < C_; c += 4) {
        float s = 0.f, s2 = 0.f;
#pragma unroll
        for (int d = lane; d < D_; d += 32) {
            const float v = sh_h[c * SH + d];
            s += v; s2 += v * v;
        }
#pragma unroll
        for (int o = 16; o > 0; o >>= 1) {
            s  += __shfl_xor(s,  o, 32);
            s2 += __shfl_xor(s2, o, 32);
        }
        const float mu = s * (1.0f / D_);
        const float rs = rsqrtf(s2 * (1.0f / D_) - mu * mu + 1e-5f);
#pragma unroll
        for (int d = lane; d < D_; d += 32) {
            const float v = (sh_h[c * SH + d] - mu) * rs * ln_w[d] + ln_b[d];
            sh_hn[c * SN + d] = (half_t)v;
        }
    }
    // zero pad rows 21..31 with 32-bit writes (two halves per store)
    for (int idx = t; idx < (32 - C_) * (D_ / 2); idx += BDIM) {
        const int c = C_ + idx / (D_ / 2), w = idx % (D_ / 2);
        ((unsigned int*)(sh_hn + c * SN))[w] = 0u;
    }
    __syncthreads();

    // ---- 3. Stage A: h_low = h_norm[32x256] @ WdT' + bd  (1 tile per wave) --
    {
        const int mt = wave >> 1, nt = wave & 1;
        v8f acc = {0.f, 0.f, 0.f, 0.f, 0.f, 0.f, 0.f, 0.f};
#pragma unroll
        for (int k0 = 0; k0 < D_; k0 += 32) {
            v16h a  = load_a_lds(sh_hn, mt * 16, k0, SN);
            v16h bf = load_b_glb(WdT, nt * 16, k0, D_);
            acc = wmma_f16(a, bf, acc);
        }
        const int mbase = mt * 16 + ((lane >> 4) ? 8 : 0);
        const int n     = nt * 16 + (lane & 15);
        const float bias = bd[n];
#pragma unroll
        for (int r = 0; r < 8; ++r) sh_hlow[(mbase + r) * 33 + n] = acc[r] + bias;
    }
    __syncthreads();

    // ---- 4. h_low -> f16 A-buffer; channel variance (unbiased over C) ------
    for (int idx = t; idx < 32 * R_; idx += BDIM) {
        const int c = idx >> 5, r = idx & 31;
        sh_hlow16[c * 34 + r] = (half_t)((c < C_) ? sh_hlow[c * 33 + r] : 0.0f);
    }
    for (int d = t; d < D_; d += BDIM) {
        float s = 0.f, s2 = 0.f;
#pragma unroll
        for (int c = 0; c < C_; ++c) {
            const float v = sh_h[c * SH + d];
            s += v; s2 += v * v;
        }
        const float m = s * (1.0f / C_);
        sh_var[d] = (s2 - s * m) * (1.0f / (C_ - 1));
    }
    __syncthreads();
    // data MLP hidden: gelu(var @ dn1 + dn1_b)
    if (t < H_) {
        float a = dn1b[t];
#pragma unroll 8
        for (int d = 0; d < D_; ++d) a += sh_var[d] * dn1w[d * H_ + t];
        sh_hid[t] = gelu_exact(a);
    }
    __syncthreads();
    // adaptive gate: sigmoid(channel_logits + eps * (hid @ dn2 + dn2_b))
    {
        const float eps = *eps_p;
        for (int d = t; d < D_; d += BDIM) {
            float a = dn2b[d];
#pragma unroll 8
            for (int j = 0; j < H_; ++j) a += sh_hid[j] * dn2w[j * D_ + d];
            sh_gate[d] = sigm(chl[d] + eps * a);
        }
    }
    __syncthreads();

    // ---- 5. Stage B: Q/K/V/G = h_low16 @ W*T'  (one product per wave) ------
    {
        const half_t* Wt = (wave == 0) ? WqT : (wave == 1) ? WkT : (wave == 2) ? WvT : WgT;
        float* dst = sh_qkvg + wave * (32 * 33);
#pragma unroll
        for (int mt = 0; mt < 2; ++mt) {
#pragma unroll
            for (int nt = 0; nt < 2; ++nt) {
                v8f acc = {0.f, 0.f, 0.f, 0.f, 0.f, 0.f, 0.f, 0.f};
                v16h a  = load_a_lds(sh_hlow16, mt * 16, 0, 34);
                v16h bf = load_b_glb(Wt, nt * 16, 0, R_);
                acc = wmma_f16(a, bf, acc);
                const int mbase = mt * 16 + ((lane >> 4) ? 8 : 0);
                const int n     = nt * 16 + (lane & 15);
#pragma unroll
                for (int r = 0; r < 8; ++r) dst[(mbase + r) * 33 + n] = acc[r];
            }
        }
    }
    __syncthreads();

    // ---- 6. l2norm rows, global pool over C, content gate -> A2 (f16) ------
    float* Qm = sh_qkvg;
    float* Km = sh_qkvg + 1 * 32 * 33;
    float* Vm = sh_qkvg + 2 * 32 * 33;
    float* Gm = sh_qkvg + 3 * 32 * 33;
    if (t < C_) {
        float sq = 0.f, sk = 0.f;
#pragma unroll
        for (int r = 0; r < R_; ++r) {
            const float q = Qm[t * 33 + r], k = Km[t * 33 + r];
            sq += q * q; sk += k * k;
        }
        sh_qinv[t] = 1.0f / fmaxf(sqrtf(sq), 1e-12f);
        sh_kinv[t] = 1.0f / fmaxf(sqrtf(sk), 1e-12f);
    }
    __syncthreads();
    if (t >= 32 && t < 64) {
        const int r = t - 32;
        float s = 0.f;
#pragma unroll
        for (int c = 0; c < C_; ++c) s += Km[c * 33 + r] * sh_kinv[c] * Vm[c * 33 + r];
        sh_gf[r] = s;
    }
    __syncthreads();
    for (int idx = t; idx < 32 * 32; idx += BDIM) {
        const int c = idx >> 5, r = idx & 31;
        float v = 0.0f;
        if (c < C_) {
            const float g = sigm(Gm[c * 33 + r] + bg[r]);
            v = Qm[c * 33 + r] * sh_qinv[c] * sh_gf[r] * g;
        }
        sh_a2[c * 34 + r] = (half_t)v;
    }
    __syncthreads();

    // ---- 7. Stage C: mixed = A2 @ WuT' + bu; fused residual + gate + store -
    for (int nt = wave * 4; nt < wave * 4 + 4; ++nt) {
#pragma unroll
        for (int mt = 0; mt < 2; ++mt) {
            v8f acc = {0.f, 0.f, 0.f, 0.f, 0.f, 0.f, 0.f, 0.f};
            v16h a  = load_a_lds(sh_a2, mt * 16, 0, 34);
            v16h bf = load_b_glb(WuT, nt * 16, 0, R_);
            acc = wmma_f16(a, bf, acc);
            const int mbase = mt * 16 + ((lane >> 4) ? 8 : 0);
            const int d     = nt * 16 + (lane & 15);
            const float bias = bu[d];
            const float g    = sh_gate[d];
#pragma unroll
            for (int r = 0; r < 8; ++r) {
                const int c = mbase + r;
                if (c < C_) {
                    const float res = sh_h[c * SH + d] + g * (acc[r] + bias);
                    out[((size_t)(b * C_ + c) * P_ + p) * D_ + d] = res;
                }
            }
        }
    }
}

// ---------------------------------------------------------------------------
extern "C" void kernel_launch(void* const* d_in, const int* in_sizes, int n_in,
                              void* d_out, int out_size, void* d_ws, size_t ws_size,
                              hipStream_t stream) {
    const float* x    = (const float*)d_in[0];
    const float* ln_w = (const float*)d_in[1];
    const float* ln_b = (const float*)d_in[2];
    const float* Wd   = (const float*)d_in[3];
    const float* bd   = (const float*)d_in[4];
    const float* Wq   = (const float*)d_in[5];
    const float* Wk   = (const float*)d_in[6];
    const float* Wv   = (const float*)d_in[7];
    const float* Wg   = (const float*)d_in[8];
    const float* bg   = (const float*)d_in[9];
    const float* Wu   = (const float*)d_in[10];
    const float* bu   = (const float*)d_in[11];
    const float* cn1w = (const float*)d_in[12];
    const float* cn1b = (const float*)d_in[13];
    const float* cn2w = (const float*)d_in[14];
    const float* cn2b = (const float*)d_in[15];
    const float* dn1w = (const float*)d_in[16];
    const float* dn1b = (const float*)d_in[17];
    const float* dn2w = (const float*)d_in[18];
    const float* dn2b = (const float*)d_in[19];
    const float* eps  = (const float*)d_in[20];
    char* ws   = (char*)d_ws;
    float* out = (float*)d_out;

    hydra_prep<<<1, 256, 0, stream>>>(Wd, Wq, Wk, Wv, Wg, Wu, cn1w, cn1b, cn2w, cn2b, ws);
    hydra_main<<<64 * P_, BDIM, 0, stream>>>(x, ln_w, ln_b, bd, bg, bu,
                                             dn1w, dn1b, dn2w, dn2b, eps,
                                             (const char*)ws, out);
}
